// GAT_80470507257934
// MI455X (gfx1250) — matrix operand order repaired and checked by reading, not compiled, not run on hardware
//
#include <hip/hip_runtime.h>
#include <hip/hip_bf16.h>
#include <math.h>

// ---------- types ----------
typedef _Float16 v16h __attribute__((ext_vector_type(16)));
typedef _Float16 h8   __attribute__((ext_vector_type(8)));
typedef float    v8f  __attribute__((ext_vector_type(8)));
typedef float    fl4  __attribute__((ext_vector_type(4)));
typedef float    fl2  __attribute__((ext_vector_type(2)));

union FragAB { v16h v; h8 q[2]; };
union FragC  { v8f  v; float e[8]; };

// ---------- small utility kernels ----------
__global__ void k_zero_i32(int* p, int n) {
  int i = blockIdx.x * blockDim.x + threadIdx.x;
  if (i < n) p[i] = 0;
}
__global__ void k_copy_i32(const int* __restrict__ a, int* __restrict__ b, int n) {
  int i = blockIdx.x * blockDim.x + threadIdx.x;
  if (i < n) b[i] = a[i];
}
__global__ void k_f32_to_f16(const float* __restrict__ in, _Float16* __restrict__ out, int n) {
  int i = blockIdx.x * blockDim.x + threadIdx.x;
  if (i < n) out[i] = (_Float16)in[i];
}
// Wt[ncol][k] = (f16) W[k][ncol]   (W is [K, Ncols] row major)
__global__ void k_transpose_f16(const float* __restrict__ W, _Float16* __restrict__ Wt,
                                int K, int Ncols) {
  int i = blockIdx.x * blockDim.x + threadIdx.x;
  if (i >= K * Ncols) return;
  int n = i / K, k = i % K;
  Wt[(size_t)n * K + k] = (_Float16)W[(size_t)k * Ncols + n];
}

// ---------- CSR build ----------
__global__ void k_hist(const int* __restrict__ dst, int* __restrict__ cnt, int E) {
  int e = blockIdx.x * blockDim.x + threadIdx.x;
  if (e < E) atomicAdd(&cnt[dst[e]], 1);
}
// single-block exclusive scan of cnt[0..N) -> row_ptr[0..N]
__global__ void k_scan(const int* __restrict__ cnt, int* __restrict__ row_ptr, int N) {
  __shared__ int buf[1024];
  __shared__ int carry;
  int t = threadIdx.x;
  if (t == 0) carry = 0;
  __syncthreads();
  for (int base = 0; base < N; base += 1024) {
    int i = base + t;
    int v = (i < N) ? cnt[i] : 0;
    buf[t] = v;
    __syncthreads();
    for (int off = 1; off < 1024; off <<= 1) {
      int x = (t >= off) ? buf[t - off] : 0;
      __syncthreads();
      buf[t] += x;
      __syncthreads();
    }
    if (i < N) row_ptr[i] = carry + buf[t] - v;
    __syncthreads();
    if (t == 0) carry += buf[1023];
    __syncthreads();
  }
  if (t == 0) row_ptr[N] = carry;
}
__global__ void k_fill(const int* __restrict__ dst, int* __restrict__ cursor,
                       int* __restrict__ edge_ids, int E) {
  int e = blockIdx.x * blockDim.x + threadIdx.x;
  if (e >= E) return;
  int pos = atomicAdd(&cursor[dst[e]], 1);
  edge_ids[pos] = e;
}
// per-node insertion sort -> deterministic edge order (degree ~16)
__global__ void k_sort(const int* __restrict__ row_ptr, int* __restrict__ edge_ids, int N) {
  int n = blockIdx.x * blockDim.x + threadIdx.x;
  if (n >= N) return;
  int s = row_ptr[n], e = row_ptr[n + 1];
  for (int i = s + 1; i < e; ++i) {
    int v = edge_ids[i], j = i - 1;
    while (j >= s && edge_ids[j] > v) { edge_ids[j + 1] = edge_ids[j]; --j; }
    edge_ids[j + 1] = v;
  }
}

// ---------- WMMA GEMM: C[M,Ncols] = A[M,K] (f16) * B[K,Ncols] (f16, given as Bt[Ncols,K]) ----------
// Register-blocked: one wave computes a 16x64 output strip. Per k-step the A
// fragment is loaded once and reused against 4 B fragments (4 WMMAs / 10 b128
// loads). A and B both live in the 192MB L2, so no LDS staging is needed.
__global__ void k_gemm_wmma(const _Float16* __restrict__ A,
                            const _Float16* __restrict__ Bt,
                            float* __restrict__ C,
                            int M, int K, int Ncols) {
  const int lane = threadIdx.x & 31;
  const int wv = (blockIdx.x * blockDim.x + threadIdx.x) >> 5;
  const int tiles_n = Ncols >> 6;   // 64-wide column strips
  const int tiles_m = M >> 4;
  if (wv >= tiles_m * tiles_n) return;
  const int tm = wv / tiles_n, tn = wv % tiles_n;
  const int l15 = lane & 15;
  const int g = lane >> 4;

  const _Float16* arow = A + (size_t)(tm * 16 + l15) * K;
  // B fragment j covers columns tn*64 + j*16 + [0..15]
  const _Float16* brow = Bt + (size_t)(tn * 64 + l15) * K;

  v8f acc[4];
#pragma unroll
  for (int j = 0; j < 4; ++j) acc[j] = (v8f){0.f, 0.f, 0.f, 0.f, 0.f, 0.f, 0.f, 0.f};

  for (int k0 = 0; k0 < K; k0 += 32) {
    FragAB a;
    // A (16-bit, 16x32): lane holds K in [k0+8g, k0+8g+8) U [k0+16+8g, k0+16+8g+8)
    a.q[0] = *(const h8*)(arow + k0 + 8 * g);
    a.q[1] = *(const h8*)(arow + k0 + 16 + 8 * g);
#pragma unroll
    for (int j = 0; j < 4; ++j) {
      FragAB b;
      // B (16-bit, 32x16): lane n holds K in [k0+8g, k0+8g+8) U [k0+16+8g, k0+16+8g+8)
      const _Float16* bp = brow + (size_t)j * 16 * K + k0 + 8 * g;
      b.q[0] = *(const h8*)(bp);
      b.q[1] = *(const h8*)(bp + 16);
      acc[j] = __builtin_amdgcn_wmma_f32_16x16x32_f16(false, a.v, false, b.v,
                                                      (short)0, acc[j], false, false);
    }
  }
  // C/D layout: VGPR r -> row (r + 8g), col l15
#pragma unroll
  for (int j = 0; j < 4; ++j) {
    FragC cu; cu.v = acc[j];
    float* crow = C + (size_t)(tm * 16) * Ncols + tn * 64 + j * 16 + l15;
#pragma unroll
    for (int r = 0; r < 8; ++r)
      crow[(size_t)(r + 8 * g) * Ncols] = cu.e[r];
  }
}

// ---------- el/er: per (node, head) dot of feat with al/ar ----------
__global__ void k_eler(const float* __restrict__ feat,
                       const float* __restrict__ al, const float* __restrict__ ar,
                       float* __restrict__ el, float* __restrict__ er,
                       int N, int H, int F) {
  int idx = blockIdx.x * blockDim.x + threadIdx.x;
  if (idx >= N * H) return;
  int n = idx / H, h = idx % H;
  const float* fp = feat + (size_t)n * H * F + h * F;
  const float* ap = al + h * F;
  const float* rp = ar + h * F;
  float sl = 0.f, sr = 0.f;
  for (int f = 0; f < F; f += 4) {
    fl4 fv = *(const fl4*)(fp + f);
    fl4 av = *(const fl4*)(ap + f);
    fl4 rv = *(const fl4*)(rp + f);
    sl += fv.x * av.x + fv.y * av.y + fv.z * av.z + fv.w * av.w;
    sr += fv.x * rv.x + fv.y * rv.y + fv.z * rv.z + fv.w * rv.w;
  }
  el[idx] = sl;
  er[idx] = sr;
}

// ---------- helpers ----------
__device__ __forceinline__ fl4 lrelu4(fl4 x) {
  fl4 r;
  r.x = x.x > 0.f ? x.x : 0.2f * x.x;
  r.y = x.y > 0.f ? x.y : 0.2f * x.y;
  r.z = x.z > 0.f ? x.z : 0.2f * x.z;
  r.w = x.w > 0.f ? x.w : 0.2f * x.w;
  return r;
}
__device__ __forceinline__ fl4 exp4(fl4 x) {
  fl4 r; r.x = __expf(x.x); r.y = __expf(x.y); r.z = __expf(x.z); r.w = __expf(x.w);
  return r;
}

// ---------- per-dst-node softmax + aggregation (one wave32 per node) ----------
template <int H>
__global__ void k_agg(const float* __restrict__ feat,
                      const float* __restrict__ el,
                      const float* __restrict__ er,
                      const int* __restrict__ row_ptr,
                      const int* __restrict__ edge_ids,
                      const int* __restrict__ srcv,
                      const float* __restrict__ bias,
                      float* __restrict__ out, int N) {
  const int lane = threadIdx.x & 31;
  const int n = (blockIdx.x * blockDim.x + threadIdx.x) >> 5;
  if (n >= N) return;
  const int s0 = row_ptr[n], s1 = row_ptr[n + 1];

  if constexpr (H == 4) {
    const fl4 ern = *(const fl4*)(er + (size_t)n * 4);
    // pass 1: per-head max over incoming edges
    fl4 m = {-3.4e38f, -3.4e38f, -3.4e38f, -3.4e38f};
    for (int i = s0 + lane; i < s1; i += 32) {
      int sn = srcv[edge_ids[i]];
      fl4 x = lrelu4(*(const fl4*)(el + (size_t)sn * 4) + ern);
      m.x = fmaxf(m.x, x.x); m.y = fmaxf(m.y, x.y);
      m.z = fmaxf(m.z, x.z); m.w = fmaxf(m.w, x.w);
    }
    for (int off = 16; off; off >>= 1) {
      m.x = fmaxf(m.x, __shfl_xor(m.x, off));
      m.y = fmaxf(m.y, __shfl_xor(m.y, off));
      m.z = fmaxf(m.z, __shfl_xor(m.z, off));
      m.w = fmaxf(m.w, __shfl_xor(m.w, off));
    }
    // pass 2: per-head sum of exp
    fl4 ssum = {0.f, 0.f, 0.f, 0.f};
    for (int i = s0 + lane; i < s1; i += 32) {
      int sn = srcv[edge_ids[i]];
      fl4 x = lrelu4(*(const fl4*)(el + (size_t)sn * 4) + ern);
      ssum += exp4(x - m);
    }
    for (int off = 16; off; off >>= 1) {
      ssum.x += __shfl_xor(ssum.x, off);
      ssum.y += __shfl_xor(ssum.y, off);
      ssum.z += __shfl_xor(ssum.z, off);
      ssum.w += __shfl_xor(ssum.w, off);
    }
    fl4 rs;
    rs.x = 1.f / fmaxf(ssum.x, 1e-9f);
    rs.y = 1.f / fmaxf(ssum.y, 1e-9f);
    rs.z = 1.f / fmaxf(ssum.z, 1e-9f);
    rs.w = 1.f / fmaxf(ssum.w, 1e-9f);
    // pass 3: weighted feature gather; lanes cover all 256 features.
    // Prefetch the next edge's feat row to hide random-gather latency
    // (emits global_prefetch_b8 on gfx1250).
    const int c0 = lane * 4;        // heads 0/1
    const int c1 = 128 + lane * 4;  // heads 2/3
    const int h0 = lane >> 4, h2 = h0 + 2;
    fl4 acc0 = {0.f, 0.f, 0.f, 0.f}, acc1 = {0.f, 0.f, 0.f, 0.f};
    for (int i = s0; i < s1; ++i) {
      int sn = srcv[edge_ids[i]];
      if (i + 1 < s1) {
        int sn_next = srcv[edge_ids[i + 1]];
        __builtin_prefetch(feat + (size_t)sn_next * 256 + c0, 0, 1);
        __builtin_prefetch(feat + (size_t)sn_next * 256 + c1, 0, 1);
      }
      fl4 x = lrelu4(*(const fl4*)(el + (size_t)sn * 4) + ern);
      fl4 a = exp4(x - m) * rs;
      float w0 = a[h0], w1 = a[h2];
      const float* fp = feat + (size_t)sn * 256;
      acc0 += w0 * *(const fl4*)(fp + c0);
      acc1 += w1 * *(const fl4*)(fp + c1);
    }
    fl4 r0 = acc0 + *(const fl4*)(bias + c0);
    fl4 r1 = acc1 + *(const fl4*)(bias + c1);
    *(fl4*)(out + (size_t)n * 256 + c0) = r0;
    *(fl4*)(out + (size_t)n * 256 + c1) = r1;
  } else {  // H == 1, F == 64
    const float ern = er[n];
    float m = -3.4e38f;
    for (int i = s0 + lane; i < s1; i += 32) {
      int sn = srcv[edge_ids[i]];
      float x = el[sn] + ern; x = x > 0.f ? x : 0.2f * x;
      m = fmaxf(m, x);
    }
    for (int off = 16; off; off >>= 1) m = fmaxf(m, __shfl_xor(m, off));
    float ssum = 0.f;
    for (int i = s0 + lane; i < s1; i += 32) {
      int sn = srcv[edge_ids[i]];
      float x = el[sn] + ern; x = x > 0.f ? x : 0.2f * x;
      ssum += __expf(x - m);
    }
    for (int off = 16; off; off >>= 1) ssum += __shfl_xor(ssum, off);
    float rs = 1.f / fmaxf(ssum, 1e-9f);
    const int c = lane * 2;
    fl2 acc = {0.f, 0.f};
    for (int i = s0; i < s1; ++i) {
      int sn = srcv[edge_ids[i]];
      if (i + 1 < s1) {
        int sn_next = srcv[edge_ids[i + 1]];
        __builtin_prefetch(feat + (size_t)sn_next * 64 + c, 0, 1);
      }
      float x = el[sn] + ern; x = x > 0.f ? x : 0.2f * x;
      float w = __expf(x - m) * rs;
      acc += w * *(const fl2*)(feat + (size_t)sn * 64 + c);
    }
    fl2 r = acc + *(const fl2*)(bias + c);
    *(fl2*)(out + (size_t)n * 64 + c) = r;
  }
}

// ---------- host side ----------
extern "C" void kernel_launch(void* const* d_in, const int* in_sizes, int n_in,
                              void* d_out, int out_size, void* d_ws, size_t ws_size,
                              hipStream_t stream) {
  const float* h0   = (const float*)d_in[0];
  const int*   src  = (const int*)d_in[1];
  const int*   dst  = (const int*)d_in[2];
  const int*   ids  = (const int*)d_in[3];
  const float* W0   = (const float*)d_in[4];
  const float* al0  = (const float*)d_in[5];
  const float* ar0  = (const float*)d_in[6];
  const float* b0   = (const float*)d_in[7];
  const float* W1   = (const float*)d_in[8];
  const float* al1  = (const float*)d_in[9];
  const float* ar1  = (const float*)d_in[10];
  const float* b1   = (const float*)d_in[11];
  const float* W2   = (const float*)d_in[12];
  const float* al2  = (const float*)d_in[13];
  const float* ar2  = (const float*)d_in[14];
  const float* b2   = (const float*)d_in[15];

  const int Din = 256, HF = 256, H = 4, F = 64, Dout = 64;
  const int N  = in_sizes[0] / Din;
  const int E  = in_sizes[1];
  const int NQ = in_sizes[3];

  // workspace carve-up (bump allocator, 256B aligned)
  char* base = (char*)d_ws;
  size_t off = 0;
  auto take = [&](size_t bytes) -> void* {
    void* p = base + off;
    off = (off + bytes + 255) & ~(size_t)255;
    return p;
  };
  _Float16* f16A     = (_Float16*)take((size_t)N * 256 * sizeof(_Float16));
  _Float16* Wt16     = (_Float16*)take((size_t)256 * 256 * sizeof(_Float16));
  float*    feat     = (float*)take((size_t)N * 256 * sizeof(float));
  float*    hbuf     = (float*)take((size_t)N * 256 * sizeof(float));
  float*    el       = (float*)take((size_t)N * H * sizeof(float));
  float*    er       = (float*)take((size_t)N * H * sizeof(float));
  int*      row_ptr  = (int*)take((size_t)(N + 1) * sizeof(int));
  int*      cursor   = (int*)take((size_t)N * sizeof(int));
  int*      edge_ids = (int*)take((size_t)E * sizeof(int));
  (void)ws_size;

  const int TB = 256;
  auto blk = [](long long n, int t) { return (int)((n + t - 1) / t); };

  // ---- CSR by dst (deterministic: per-node sorted edge lists) ----
  k_zero_i32<<<blk(N, TB), TB, 0, stream>>>(cursor, N);          // reuse as counts
  k_hist<<<blk(E, TB), TB, 0, stream>>>(dst, cursor, E);
  k_scan<<<1, 1024, 0, stream>>>(cursor, row_ptr, N);
  k_copy_i32<<<blk(N, TB), TB, 0, stream>>>(row_ptr, cursor, N);
  k_fill<<<blk(E, TB), TB, 0, stream>>>(dst, cursor, edge_ids, E);
  k_sort<<<blk(N, TB), TB, 0, stream>>>(row_ptr, edge_ids, N);

  // ---- one GAT layer ----
  auto layer = [&](const float* in_h, const float* W, int Ncols,
                   const float* al, const float* ar, const float* bias,
                   int Hh, float* out_h) {
    const int Ff = Ncols / Hh;
    k_f32_to_f16<<<blk((long long)N * 256, TB), TB, 0, stream>>>(in_h, f16A, N * 256);
    k_transpose_f16<<<blk((long long)256 * Ncols, TB), TB, 0, stream>>>(W, Wt16, 256, Ncols);
    long long waves = (long long)(N >> 4) * (Ncols >> 6);  // 16x64 strips
    k_gemm_wmma<<<blk(waves, 8), TB, 0, stream>>>(f16A, Wt16, feat, N, 256, Ncols);
    k_eler<<<blk((long long)N * Hh, TB), TB, 0, stream>>>(feat, al, ar, el, er, N, Hh, Ff);
    if (Hh == 4)
      k_agg<4><<<blk((long long)N * 32, TB), TB, 0, stream>>>(feat, el, er, row_ptr, edge_ids,
                                                              src, bias, out_h, N);
    else
      k_agg<1><<<blk((long long)N * 32, TB), TB, 0, stream>>>(feat, el, er, row_ptr, edge_ids,
                                                              src, bias, out_h, N);
  };

  float* out_f = (float*)d_out;
  layer(h0,   W0, HF,   al0, ar0, b0, H, hbuf);   // layer 0: 256 -> 4x64
  layer(hbuf, W1, HF,   al1, ar1, b1, H, hbuf);   // layer 1: 256 -> 4x64 (agg reads feat, not hbuf)
  layer(hbuf, W2, Dout, al2, ar2, b2, 1, out_f);  // layer 2: 256 -> 1x64 -> d_out

  // append dst_node_ids (int bits) after the N*64 float output
  k_copy_i32<<<blk(NQ, TB), TB, 0, stream>>>(ids, (int*)(out_f + (size_t)N * Dout), NQ);
}
